// Attention_19404662243663
// MI455X (gfx1250) — matrix operand order repaired and checked by reading, not compiled, not run on hardware
//
#include <hip/hip_runtime.h>

// ---------------------------------------------------------------------------
// MHA forward for MI455X (gfx1250, wave32, WMMA).
// S=2048, B=2, H=1024, NH=16, DK=64.
// All matmuls use v_wmma_f32_16x16x32_bf16 with f32 accumulation.
// Attention is flash-style: row stats pass + fused (scores->softmax->att
// write + ctx GEMM) pass, so the 0.54 GB att tensor is written exactly once
// (non-temporal) and never re-read.
// ---------------------------------------------------------------------------

#define S_LEN 2048
#define BB    2
#define H_DIM 1024
#define NHEAD 16
#define DKD   64
#define MROWS (S_LEN * BB)          // 4096 rows for [S,B,*] GEMMs

typedef __attribute__((ext_vector_type(16))) __bf16        v16bf;
typedef __attribute__((ext_vector_type(8)))  float         v8f;
typedef __attribute__((ext_vector_type(8)))  unsigned int  v8u;

// ----- bf16 helpers (round-to-nearest-even) --------------------------------
__device__ __forceinline__ unsigned int f32_to_bf16_bits(float f) {
  unsigned int x = __builtin_bit_cast(unsigned int, f);
  return (x + 0x7FFFu + ((x >> 16) & 1u)) >> 16;
}

// ----- WMMA fragment loaders (layouts per CDNA5 ISA §7.12.2) ----
// A 16x32 bf16: lanes 0-15 row m hold K = 0..7,16..23 ; lanes 16-31 hold
// K = 8..15,24..31.  Source is row-major with K contiguous (ld elements/row).
__device__ __forceinline__ v16bf load_a_bf16(const unsigned short* p, int ld, int lane) {
  int row = lane & 15;
  int kb  = (lane >> 4) << 3;                 // 0 or 8
  const unsigned short* q = p + (size_t)row * ld + kb;
  v8u u;
  u[0] = *(const unsigned int*)(q + 0);
  u[1] = *(const unsigned int*)(q + 2);
  u[2] = *(const unsigned int*)(q + 4);
  u[3] = *(const unsigned int*)(q + 6);
  u[4] = *(const unsigned int*)(q + 16);
  u[5] = *(const unsigned int*)(q + 18);
  u[6] = *(const unsigned int*)(q + 20);
  u[7] = *(const unsigned int*)(q + 22);
  return __builtin_bit_cast(v16bf, u);
}

// B 32x16 bf16: lane n = column; lanes 0-15 hold K=0..7,16..23, lanes 16-31 hold
// K=8..15,24..31 (two per VGPR).  Source rows = output columns, K contiguous.
__device__ __forceinline__ v16bf load_b_bf16(const unsigned short* p, int ld, int lane) {
  int col = lane & 15;
  int kb  = (lane >> 4) << 3;                 // 0 or 8
  const unsigned short* q = p + (size_t)col * ld + kb;
  v8u u;
#pragma unroll
  for (int i = 0; i < 8; ++i) u[i] = *(const unsigned int*)(q + 2 * i + ((i >> 2) << 3));
  return __builtin_bit_cast(v16bf, u);
}

__device__ __forceinline__ v8f wmma_bf16(v16bf a, v16bf b, v8f c) {
  return __builtin_amdgcn_wmma_f32_16x16x32_bf16(false, a, false, b, (short)0, c,
                                                 false, false);
}

// ---------------------------------------------------------------------------
// 1) f32 -> bf16 conversion (grid-stride)
// ---------------------------------------------------------------------------
__global__ void cvt_bf16_kernel(const float* __restrict__ src,
                                unsigned short* __restrict__ dst, int n) {
  int i = blockIdx.x * blockDim.x + threadIdx.x;
  int stride = gridDim.x * blockDim.x;
  for (; i < n; i += stride) dst[i] = (unsigned short)f32_to_bf16_bits(src[i]);
}

// ---------------------------------------------------------------------------
// 2) QKV projection: D[m, n'] = x[m, :] . W[n', :] + b[n']
//    m = s*B + b ; n' = head*64 + d.
//    Q,K stored bf16 [head][b][s][d]; V stored TRANSPOSED bf16 [head][b][d][s].
//    grid (MROWS/128, H/64, 3), block 128 (4 waves, each 32x64 tile)
// ---------------------------------------------------------------------------
__global__ void qkv_kernel(const unsigned short* __restrict__ xb,
                           const unsigned short* __restrict__ wq,
                           const unsigned short* __restrict__ wk,
                           const unsigned short* __restrict__ wv,
                           const float* __restrict__ bq,
                           const float* __restrict__ bk,
                           const float* __restrict__ bv,
                           unsigned short* __restrict__ Qb,
                           unsigned short* __restrict__ Kb,
                           unsigned short* __restrict__ Vtb) {
  int lane = threadIdx.x & 31;
  int wave = threadIdx.x >> 5;
  int proj = blockIdx.z;
  const unsigned short* w = (proj == 0) ? wq : ((proj == 1) ? wk : wv);
  const float* bias = (proj == 0) ? bq : ((proj == 1) ? bk : bv);

  int m0 = blockIdx.x * 128 + wave * 32;
  int n0 = blockIdx.y * 64;

  v8f acc[8];
#pragma unroll
  for (int i = 0; i < 8; ++i) acc[i] = (v8f){};

  for (int k0 = 0; k0 < H_DIM; k0 += 32) {
    v16bf a0 = load_a_bf16(xb + (size_t)m0 * H_DIM + k0, H_DIM, lane);
    v16bf a1 = load_a_bf16(xb + (size_t)(m0 + 16) * H_DIM + k0, H_DIM, lane);
#pragma unroll
    for (int j = 0; j < 4; ++j) {
      v16bf bf = load_b_bf16(w + (size_t)(n0 + 16 * j) * H_DIM + k0, H_DIM, lane);
      acc[2 * j]     = wmma_bf16(a0, bf, acc[2 * j]);
      acc[2 * j + 1] = wmma_bf16(a1, bf, acc[2 * j + 1]);
    }
  }

  int ncol  = lane & 15;
  int mhalf = (lane >> 4) * 8;
#pragma unroll
  for (int j = 0; j < 4; ++j) {
    int ngl = n0 + 16 * j + ncol;
    float bv_ = bias[ngl];
    int nh = ngl >> 6, d = ngl & 63;
#pragma unroll
    for (int half = 0; half < 2; ++half) {
      v8f c = acc[2 * j + half];
      int mbase = m0 + 16 * half;
#pragma unroll
      for (int r = 0; r < 8; ++r) {
        int m = mbase + mhalf + r;
        unsigned short obf = (unsigned short)f32_to_bf16_bits(c[r] + bv_);
        int s = m >> 1, b = m & 1;
        if (proj == 2) {
          Vtb[((size_t)(nh * BB + b) * DKD + d) * S_LEN + s] = obf;
        } else {
          unsigned short* dst = (proj == 0) ? Qb : Kb;
          dst[((size_t)(nh * BB + b) * S_LEN + s) * DKD + d] = obf;
        }
      }
    }
  }
}

// ---------------------------------------------------------------------------
// 3) Phase 1: per-row softmax stats (max, 1/sum) via on-the-fly scores.
//    grid (S/64, NHEAD*BB), block 128 (4 waves, each 16 query rows).
//    Online per-lane (max,sum) update; cross-lane merge with shfl_xor.
// ---------------------------------------------------------------------------
__global__ void rowstats_kernel(const unsigned short* __restrict__ Qb,
                                const unsigned short* __restrict__ Kb,
                                float* __restrict__ Rmax,
                                float* __restrict__ Rinv) {
  int lane = threadIdx.x & 31;
  int wave = threadIdx.x >> 5;
  int nb = blockIdx.y;
  int i0 = blockIdx.x * 64 + wave * 16;

  const unsigned short* q = Qb + (size_t)nb * S_LEN * DKD;
  const unsigned short* k = Kb + (size_t)nb * S_LEN * DKD;

  int ncol = lane & 15;
  int hi   = lane >> 4;

  // Q A-fragments for our 16 rows, K-dim 0..31 and 32..63 (kept all pass)
  v16bf aQ0 = load_a_bf16(q + (size_t)i0 * DKD + 0,  DKD, lane);
  v16bf aQ1 = load_a_bf16(q + (size_t)i0 * DKD + 32, DKD, lane);

  float mrow[8], srow[8];
#pragma unroll
  for (int r = 0; r < 8; ++r) { mrow[r] = -3.0e38f; srow[r] = 0.f; }

  const float scale = 0.125f;  // 1/sqrt(64)

  for (int j0 = 0; j0 < S_LEN; j0 += 32) {
    v8f s0 = (v8f){}, s1 = (v8f){};
    {
      v16bf b00 = load_b_bf16(k + (size_t)j0 * DKD + 0,  DKD, lane);
      v16bf b01 = load_b_bf16(k + (size_t)j0 * DKD + 32, DKD, lane);
      v16bf b10 = load_b_bf16(k + (size_t)(j0 + 16) * DKD + 0,  DKD, lane);
      v16bf b11 = load_b_bf16(k + (size_t)(j0 + 16) * DKD + 32, DKD, lane);
      s0 = wmma_bf16(aQ0, b00, s0);
      s0 = wmma_bf16(aQ1, b01, s0);
      s1 = wmma_bf16(aQ0, b10, s1);
      s1 = wmma_bf16(aQ1, b11, s1);
    }
#pragma unroll
    for (int r = 0; r < 8; ++r) {
      float v0 = s0[r] * scale;
      float v1 = s1[r] * scale;
      float mn = fmaxf(mrow[r], fmaxf(v0, v1));
      srow[r] = srow[r] * __expf(mrow[r] - mn) + __expf(v0 - mn) + __expf(v1 - mn);
      mrow[r] = mn;
    }
  }

  // merge the 16 column-lanes of each half-wave (masks stay within halves)
#pragma unroll
  for (int mask = 1; mask <= 8; mask <<= 1) {
#pragma unroll
    for (int r = 0; r < 8; ++r) {
      float mo = __shfl_xor(mrow[r], mask, 32);
      float so = __shfl_xor(srow[r], mask, 32);
      float mn = fmaxf(mrow[r], mo);
      srow[r] = srow[r] * __expf(mrow[r] - mn) + so * __expf(mo - mn);
      mrow[r] = mn;
    }
  }

  if (ncol == 0) {
#pragma unroll
    for (int r = 0; r < 8; ++r) {
      int idx = nb * S_LEN + i0 + 8 * hi + r;
      Rmax[idx] = mrow[r];
      Rinv[idx] = 1.0f / srow[r];
    }
  }
}

// ---------------------------------------------------------------------------
// 4) Phase 2: fused scores -> normalized att (NT store, written once) ->
//    ctx GEMM (att tile bounced through LDS into A-fragment layout).
//    grid (S/64, NHEAD*BB), block 128 (4 waves, each 16 rows x full DK=64).
// ---------------------------------------------------------------------------
__global__ void att_ctx_kernel(const unsigned short* __restrict__ Qb,
                               const unsigned short* __restrict__ Kb,
                               const unsigned short* __restrict__ Vtb,
                               const float* __restrict__ Rmax,
                               const float* __restrict__ Rinv,
                               float* __restrict__ att,
                               unsigned short* __restrict__ ctxb) {
  __shared__ unsigned short ebf[4][16 * 32];   // per-wave 16x32 bf16 att tile
  int lane = threadIdx.x & 31;
  int wave = threadIdx.x >> 5;
  int nb = blockIdx.y;
  int i0 = blockIdx.x * 64 + wave * 16;

  const unsigned short* q  = Qb  + (size_t)nb * S_LEN * DKD;
  const unsigned short* k  = Kb  + (size_t)nb * S_LEN * DKD;
  const unsigned short* vt = Vtb + (size_t)nb * DKD * S_LEN;
  float* attbase = att + (size_t)nb * S_LEN * S_LEN;

  int ncol = lane & 15;
  int hi   = lane >> 4;

  v16bf aQ0 = load_a_bf16(q + (size_t)i0 * DKD + 0,  DKD, lane);
  v16bf aQ1 = load_a_bf16(q + (size_t)i0 * DKD + 32, DKD, lane);

  float rmax[8], rinv[8];
#pragma unroll
  for (int r = 0; r < 8; ++r) {
    int idx = nb * S_LEN + i0 + 8 * hi + r;
    rmax[r] = Rmax[idx];
    rinv[r] = Rinv[idx];
  }

  v8f ctx[4];
#pragma unroll
  for (int jj = 0; jj < 4; ++jj) ctx[jj] = (v8f){};

  unsigned short* et = &ebf[wave][0];
  const float scale = 0.125f;

  for (int j0 = 0; j0 < S_LEN; j0 += 32) {
    v8f s0 = (v8f){}, s1 = (v8f){};
    {
      v16bf b00 = load_b_bf16(k + (size_t)j0 * DKD + 0,  DKD, lane);
      v16bf b01 = load_b_bf16(k + (size_t)j0 * DKD + 32, DKD, lane);
      v16bf b10 = load_b_bf16(k + (size_t)(j0 + 16) * DKD + 0,  DKD, lane);
      v16bf b11 = load_b_bf16(k + (size_t)(j0 + 16) * DKD + 32, DKD, lane);
      s0 = wmma_bf16(aQ0, b00, s0);
      s0 = wmma_bf16(aQ1, b01, s0);
      s1 = wmma_bf16(aQ0, b10, s1);
      s1 = wmma_bf16(aQ1, b11, s1);
    }
    // normalized attention weights; write att exactly once (non-temporal)
#pragma unroll
    for (int r = 0; r < 8; ++r) {
      int m = r + 8 * hi;
      float e0 = __expf(s0[r] * scale - rmax[r]) * rinv[r];
      float e1 = __expf(s1[r] * scale - rmax[r]) * rinv[r];
      size_t arow = (size_t)(i0 + m) * S_LEN + j0;
      __builtin_nontemporal_store(e0, attbase + arow + ncol);
      __builtin_nontemporal_store(e1, attbase + arow + 16 + ncol);
      et[m * 32 + ncol]      = (unsigned short)f32_to_bf16_bits(e0);
      et[m * 32 + 16 + ncol] = (unsigned short)f32_to_bf16_bits(e1);
    }
    asm volatile("s_wait_dscnt 0x0" ::: "memory");
    v16bf af = load_a_bf16(et, 32, lane);   // A-fragment from LDS tile
    asm volatile("" ::: "memory");
#pragma unroll
    for (int jj = 0; jj < 4; ++jj) {
      v16bf bv = load_b_bf16(vt + (size_t)(16 * jj) * S_LEN + j0, S_LEN, lane);
      ctx[jj] = wmma_bf16(af, bv, ctx[jj]);
    }
  }

  // store ctx bf16 [s*B+b, H] for the output projection
  int n_h = nb >> 1, b_ = nb & 1;
#pragma unroll
  for (int jj = 0; jj < 4; ++jj) {
    int d = 16 * jj + ncol;
#pragma unroll
    for (int r = 0; r < 8; ++r) {
      int i = i0 + 8 * hi + r;
      ctxb[(size_t)(i * 2 + b_) * H_DIM + n_h * 64 + d] =
          (unsigned short)f32_to_bf16_bits(ctx[jj][r]);
    }
  }
}

// ---------------------------------------------------------------------------
// 5) out[m, o] = ctx[m, :] . Wo[o, :] + bo[o]   (f32 into d_out)
//    grid (MROWS/128, H/64), block 128 (4 waves, each 32x64)
// ---------------------------------------------------------------------------
__global__ void outproj_kernel(const unsigned short* __restrict__ ctxb,
                               const unsigned short* __restrict__ wob,
                               const float* __restrict__ bo,
                               float* __restrict__ out) {
  int lane = threadIdx.x & 31;
  int wave = threadIdx.x >> 5;
  int m0 = blockIdx.x * 128 + wave * 32;
  int n0 = blockIdx.y * 64;

  v8f acc[8];
#pragma unroll
  for (int i = 0; i < 8; ++i) acc[i] = (v8f){};

  for (int k0 = 0; k0 < H_DIM; k0 += 32) {
    v16bf a0 = load_a_bf16(ctxb + (size_t)m0 * H_DIM + k0, H_DIM, lane);
    v16bf a1 = load_a_bf16(ctxb + (size_t)(m0 + 16) * H_DIM + k0, H_DIM, lane);
#pragma unroll
    for (int j = 0; j < 4; ++j) {
      v16bf bf = load_b_bf16(wob + (size_t)(n0 + 16 * j) * H_DIM + k0, H_DIM, lane);
      acc[2 * j]     = wmma_bf16(a0, bf, acc[2 * j]);
      acc[2 * j + 1] = wmma_bf16(a1, bf, acc[2 * j + 1]);
    }
  }

  int ncol  = lane & 15;
  int mhalf = (lane >> 4) * 8;
#pragma unroll
  for (int j = 0; j < 4; ++j) {
    int ngl = n0 + 16 * j + ncol;
    float bv_ = bo[ngl];
#pragma unroll
    for (int half = 0; half < 2; ++half) {
      v8f c = acc[2 * j + half];
      int mbase = m0 + 16 * half;
#pragma unroll
      for (int r = 0; r < 8; ++r) {
        int m = mbase + mhalf + r;
        out[(size_t)m * H_DIM + ngl] = c[r] + bv_;
      }
    }
  }
}

// ---------------------------------------------------------------------------
extern "C" void kernel_launch(void* const* d_in, const int* in_sizes, int n_in,
                              void* d_out, int out_size, void* d_ws, size_t ws_size,
                              hipStream_t stream) {
  (void)in_sizes; (void)n_in; (void)out_size; (void)ws_size;

  const float* x    = (const float*)d_in[0];
  const float* Wq_w = (const float*)d_in[1];
  const float* Wq_b = (const float*)d_in[2];
  const float* Wk_w = (const float*)d_in[3];
  const float* Wk_b = (const float*)d_in[4];
  const float* Wv_w = (const float*)d_in[5];
  const float* Wv_b = (const float*)d_in[6];
  const float* Wo_w = (const float*)d_in[7];
  const float* Wo_b = (const float*)d_in[8];

  float* out = (float*)d_out;                              // [S,B,H]
  float* att = out + (size_t)MROWS * H_DIM;                // [NH,B,S,S]

  char* ws = (char*)d_ws;
  unsigned short* xb   = (unsigned short*)ws; ws += (size_t)MROWS * H_DIM * 2;
  unsigned short* wqb  = (unsigned short*)ws; ws += (size_t)H_DIM * H_DIM * 2;
  unsigned short* wkb  = (unsigned short*)ws; ws += (size_t)H_DIM * H_DIM * 2;
  unsigned short* wvb  = (unsigned short*)ws; ws += (size_t)H_DIM * H_DIM * 2;
  unsigned short* wob  = (unsigned short*)ws; ws += (size_t)H_DIM * H_DIM * 2;
  unsigned short* Qb   = (unsigned short*)ws; ws += (size_t)NHEAD * BB * S_LEN * DKD * 2;
  unsigned short* Kb   = (unsigned short*)ws; ws += (size_t)NHEAD * BB * S_LEN * DKD * 2;
  unsigned short* Vtb  = (unsigned short*)ws; ws += (size_t)NHEAD * BB * DKD * S_LEN * 2;
  unsigned short* ctxb = (unsigned short*)ws; ws += (size_t)MROWS * H_DIM * 2;
  float* Rmax = (float*)ws; ws += (size_t)NHEAD * BB * S_LEN * 4;
  float* Rinv = (float*)ws; ws += (size_t)NHEAD * BB * S_LEN * 4;

  // 1) converts
  cvt_bf16_kernel<<<4096, 256, 0, stream>>>(x, xb, MROWS * H_DIM);
  cvt_bf16_kernel<<<2048, 256, 0, stream>>>(Wq_w, wqb, H_DIM * H_DIM);
  cvt_bf16_kernel<<<2048, 256, 0, stream>>>(Wk_w, wkb, H_DIM * H_DIM);
  cvt_bf16_kernel<<<2048, 256, 0, stream>>>(Wv_w, wvb, H_DIM * H_DIM);
  cvt_bf16_kernel<<<2048, 256, 0, stream>>>(Wo_w, wob, H_DIM * H_DIM);

  // 2) QKV projections (+bias), V transposed
  qkv_kernel<<<dim3(MROWS / 128, H_DIM / 64, 3), 128, 0, stream>>>(
      xb, wqb, wkb, wvb, Wq_b, Wk_b, Wv_b, Qb, Kb, Vtb);

  // 3) softmax row stats (max, 1/sum) without materializing scores
  rowstats_kernel<<<dim3(S_LEN / 64, NHEAD * BB), 128, 0, stream>>>(
      Qb, Kb, Rmax, Rinv);

  // 4) fused att (written once, NT) + ctx GEMM
  att_ctx_kernel<<<dim3(S_LEN / 64, NHEAD * BB), 128, 0, stream>>>(
      Qb, Kb, Vtb, Rmax, Rinv, att, ctxb);

  // 5) output projection (+bias) -> f32 d_out
  outproj_kernel<<<dim3(MROWS / 128, H_DIM / 64), 128, 0, stream>>>(
      ctxb, wob, Wo_b, out);
}
